// KANLayer_5609227289168
// MI455X (gfx1250) — matrix operand order repaired and checked by reading, not compiled, not run on hardware
//
#include <hip/hip_runtime.h>

// ---------------- problem constants ----------------
#define KDEG   3
#define NUMG   32
#define NB     35                      // NUM + K basis functions
#define IN_DIM 256
#define OUT_DIM 512
#define BATCH  2048
#define KSPLINE (IN_DIM * NB)          // 8960
#define KTOT    (KSPLINE + IN_DIM)     // 9216 = 144 * 64

typedef _Float16 half_t;
typedef __attribute__((ext_vector_type(16))) _Float16 v16h;
typedef __attribute__((ext_vector_type(8)))  _Float16 v8h;
typedef __attribute__((ext_vector_type(8)))  float    v8f;
typedef __attribute__((ext_vector_type(4)))  unsigned int v4u;
typedef __attribute__((ext_vector_type(8)))  int          v8i;
typedef __attribute__((ext_vector_type(4)))  int          v4i;
typedef unsigned int       u32;
typedef unsigned long long u64;

// ---------------- prep: A = [bases | silu(x)] in fp16 ----------------
__global__ void build_A(const float* __restrict__ x, half_t* __restrict__ A)
{
    int idx = blockIdx.x * blockDim.x + threadIdx.x;   // over BATCH*IN_DIM
    if (idx >= BATCH * IN_DIM) return;
    int b = idx / IN_DIM;
    int i = idx - b * IN_DIM;
    float xv = x[idx];

    float sil = xv / (1.0f + __expf(-xv));

    const float h  = 2.0f / (float)NUMG;        // 0.0625
    const float g0 = -1.0f - (float)KDEG * h;   // first extended knot
    float tf = (xv - g0) / h;
    int j = (int)floorf(tf);
    bool inr = (j >= 0) && (j < NUMG + 2 * KDEG);   // 38 degree-0 intervals

    // Cox-de-Boor on a uniform grid, 4-wide nonzero window.
    float Nw[4] = {0.f, 0.f, 0.f, 0.f};
    if (inr) {
        Nw[0] = 1.0f;
        #pragma unroll
        for (int p = 1; p <= KDEG; ++p) {
            float T[4];
            #pragma unroll
            for (int m = 0; m < 4; ++m) {
                if (m > p) { T[m] = 0.f; continue; }
                float left  = (m >= 1)     ? Nw[m - 1] : 0.f;
                float right = (m <= p - 1) ? Nw[m]     : 0.f;
                float gi  = g0 + (float)(j - p + m) * h;
                float gip = g0 + (float)(j + m + 1) * h;
                T[m] = ((xv - gi) * left + (gip - xv) * right) / ((float)p * h);
            }
            #pragma unroll
            for (int m = 0; m < 4; ++m) Nw[m] = T[m];
        }
    }

    half_t* Arow = A + (size_t)b * KTOT + (size_t)i * NB;
    int base = j - KDEG;
    #pragma unroll 1
    for (int k = 0; k < NB; ++k) {
        float v = 0.f;
        int m = k - base;
        if (inr && (m >= 0) && (m <= KDEG)) v = Nw[m];
        Arow[k] = (half_t)v;
    }
    A[(size_t)b * KTOT + KSPLINE + i] = (half_t)sil;
}

// ---------------- prep: W[o][kk] fp16, o-major (B of the GEMM) ----------------
__global__ void build_W(const float* __restrict__ coef,
                        const float* __restrict__ scale_base,
                        const float* __restrict__ scale_sp,
                        const float* __restrict__ mask,
                        half_t* __restrict__ Wt)
{
    int idx = blockIdx.x * blockDim.x + threadIdx.x;   // over IN_DIM*OUT_DIM
    if (idx >= IN_DIM * OUT_DIM) return;
    int i = idx / OUT_DIM;
    int o = idx - i * OUT_DIM;
    float mk  = mask[idx];
    float ssp = scale_sp[idx] * mk;
    float sbb = scale_base[idx] * mk;
    const float* c = coef + (size_t)idx * NB;
    half_t* wrow = Wt + (size_t)o * KTOT + (size_t)i * NB;
    #pragma unroll 1
    for (int k = 0; k < NB; ++k) wrow[k] = (half_t)(ssp * c[k]);
    Wt[(size_t)o * KTOT + KSPLINE + i] = (half_t)sbb;
}

// ---------------- WMMA GEMM with TDM (tensor_load_to_lds) staging ----------------
#define BM 128
#define BN 64
#define BK 64
#define LDSS 72    // halves per LDS row: 64 data + 8 pad (TDM pads 16B per 128B row)
#define PAD_IV 4   // pad_interval: after 32 DWORDs (128 B)
#define PAD_AMT 3  // pad_amount: 4 DWORDs (16 B)

// Issue a 2D TDM tile load: tile_d1 rows x tile_d0 elems (2B each), row stride
// in elements, LDS destination gets PAD_AMT padding every PAD_IV interval.
__device__ __forceinline__ void tdm_load_2d(u32 lds_byte_off, const void* gptr,
                                            u32 tile_d0, u32 tile_d1,
                                            u32 row_stride_elems)
{
    const u32 TD0 = 1u << 30, TD1 = 1u << 30;   // huge tensor dims: no OOB clip
    u64 ga = (u64)(uintptr_t)gptr;
    v4u g0;
    g0[0] = 1u;                                   // count=1 (valid), user mode
    g0[1] = lds_byte_off;                         // lds_addr
    g0[2] = (u32)(ga & 0xffffffffu);              // global_addr[31:0]
    g0[3] = (u32)((ga >> 32) & 0x01ffffffu) | (2u << 30);  // ga[56:32], type=2
    v8i g1;
    g1[0] = (int)((1u << 16) |                    // data_size = 2 bytes
                  (1u << 20) |                    // pad_enable
                  ((u32)PAD_IV << 22) | ((u32)PAD_AMT << 25));
    g1[1] = (int)((TD0 & 0xffffu) << 16);                       // tensor_dim0 lo
    g1[2] = (int)(((TD0 >> 16) & 0xffffu) | ((TD1 & 0xffffu) << 16));
    g1[3] = (int)(((TD1 >> 16) & 0xffffu) | (tile_d0 << 16));   // tile_dim0
    g1[4] = (int)(tile_d1);                                     // tile_dim1 (dim2=0)
    g1[5] = (int)(row_stride_elems);                            // dim0_stride lo
    g1[6] = 0;                                                  // stride hi, s1 lo
    g1[7] = 0;
    v4i z4 = {0, 0, 0, 0};
    v8i z8 = {0, 0, 0, 0, 0, 0, 0, 0};
    __builtin_amdgcn_tensor_load_to_lds(g0, g1, z4, z4, z8, 0);
}

__global__ __launch_bounds__(256)
void kan_gemm(const half_t* __restrict__ A, const half_t* __restrict__ W,
              float* __restrict__ out)
{
    __shared__ half_t Ash[2][BM * LDSS];   // 2 x 18432 B
    __shared__ half_t Bsh[2][BN * LDSS];   // 2 x  9216 B

    const int tid    = threadIdx.x;
    const int lane   = tid & 31;
    const int wave   = tid >> 5;        // 0..7
    const int wm     = wave & 3;        // wave M index (0..3), 32 rows each
    const int wn     = wave >> 2;       // wave N index (0..1), 32 cols each
    const int laneLo = lane & 15;
    const int laneHi = lane >> 4;

    const int mBlock = blockIdx.y * BM;
    const int nBlock = blockIdx.x * BN;

    const half_t* Ag = A + (size_t)mBlock * KTOT;
    const half_t* Wg = W + (size_t)nBlock * KTOT;

    v8f acc[2][2] = {};

    // prologue: DMA first K tile into buffer 0 (wave 0 issues; TDM ignores EXEC)
    if (wave == 0) {
        tdm_load_2d((u32)(uintptr_t)&Ash[0][0], Ag, BK, BM, KTOT);
        tdm_load_2d((u32)(uintptr_t)&Bsh[0][0], Wg, BK, BN, KTOT);
    }

    int buf = 0;
    for (int k0 = 0; k0 < KTOT; k0 += BK) {
        const int nxt = buf ^ 1;
        if (wave == 0) {
            if (k0 + BK < KTOT) {
                // overlap: DMA next tile into the other buffer, then wait for
                // the current pair (in-order per wave) leaving next in flight
                tdm_load_2d((u32)(uintptr_t)&Ash[nxt][0], Ag + k0 + BK, BK, BM, KTOT);
                tdm_load_2d((u32)(uintptr_t)&Bsh[nxt][0], Wg + k0 + BK, BK, BN, KTOT);
                __builtin_amdgcn_s_wait_tensorcnt((short)2);
            } else {
                __builtin_amdgcn_s_wait_tensorcnt((short)0);
            }
        }
        __syncthreads();   // buffer `buf` now valid for all waves

        const half_t* As = &Ash[buf][0];
        const half_t* Bs = &Bsh[buf][0];

        #pragma unroll
        for (int ks = 0; ks < 2; ++ks) {          // two 16x16x32 K sub-steps
            const int kk = ks * 32;
            // A fragments (16x32 f16): lanes 0-15 rows, K {0-7,16-23};
            // lanes 16-31 same rows, K {8-15,24-31}
            v16h afrag[2];
            #pragma unroll
            for (int sm = 0; sm < 2; ++sm) {
                int m = wm * 32 + sm * 16 + laneLo;
                int koff = kk + laneHi * 8;
                v8h lo = *(const v8h*)(&As[m * LDSS + koff]);
                v8h hi = *(const v8h*)(&As[m * LDSS + koff + 16]);
                #pragma unroll
                for (int e = 0; e < 8; ++e) { afrag[sm][e] = lo[e]; afrag[sm][e + 8] = hi[e]; }
            }
            // B fragments (32x16 f16): lane = N column, K {0-7,16-23} / {8-15,24-31} per lane half
            v16h bfrag[2];
            #pragma unroll
            for (int sn = 0; sn < 2; ++sn) {
                int c = wn * 32 + sn * 16 + laneLo;
                int koff = kk + laneHi * 8;
                v8h lo = *(const v8h*)(&Bs[c * LDSS + koff]);
                v8h hi = *(const v8h*)(&Bs[c * LDSS + koff + 16]);
                #pragma unroll
                for (int e = 0; e < 8; ++e) { bfrag[sn][e] = lo[e]; bfrag[sn][e + 8] = hi[e]; }
            }
            #pragma unroll
            for (int sm = 0; sm < 2; ++sm)
                #pragma unroll
                for (int sn = 0; sn < 2; ++sn)
                    acc[sm][sn] = __builtin_amdgcn_wmma_f32_16x16x32_f16(
                        false, afrag[sm], false, bfrag[sn],
                        (short)0, acc[sm][sn], false, false);
        }

        __syncthreads();   // all waves done reading `buf` before it is re-DMA'd
        buf = nxt;
    }

    // ---- epilogue: C/D layout — element r -> row (8*laneHi + r), col laneLo
    #pragma unroll
    for (int sm = 0; sm < 2; ++sm) {
        #pragma unroll
        for (int sn = 0; sn < 2; ++sn) {
            int colg = nBlock + wn * 32 + sn * 16 + laneLo;
            int rowb = mBlock + wm * 32 + sm * 16 + laneHi * 8;
            #pragma unroll
            for (int r = 0; r < 8; ++r)
                out[(size_t)(rowb + r) * OUT_DIM + colg] = acc[sm][sn][r];
        }
    }
}

// ---------------- launcher ----------------
extern "C" void kernel_launch(void* const* d_in, const int* in_sizes, int n_in,
                              void* d_out, int out_size, void* d_ws, size_t ws_size,
                              hipStream_t stream)
{
    (void)in_sizes; (void)n_in; (void)out_size; (void)ws_size;
    const float* x    = (const float*)d_in[0];
    // d_in[1] = grid: uniform knots, reconstructed analytically in build_A
    const float* coef = (const float*)d_in[2];
    const float* sb   = (const float*)d_in[3];
    const float* ssp  = (const float*)d_in[4];
    const float* mask = (const float*)d_in[5];

    half_t* A  = (half_t*)d_ws;                           // 2048*9216 fp16 = 37.75 MB
    half_t* Wt = A + (size_t)BATCH * KTOT;                // 512*9216 fp16  =  9.44 MB

    build_A<<<(BATCH * IN_DIM + 255) / 256, 256, 0, stream>>>(x, A);
    build_W<<<(IN_DIM * OUT_DIM + 255) / 256, 256, 0, stream>>>(coef, sb, ssp, mask, Wt);

    dim3 g(OUT_DIM / BN, BATCH / BM);   // (8, 16) = 128 blocks of 8 waves
    kan_gemm<<<g, 256, 0, stream>>>(A, Wt, (float*)d_out);
}
